// TMAHead_75565654606019
// MI455X (gfx1250) — compile-verified, matrix operand order checked by reading, not run on hardware
//
#include <hip/hip_runtime.h>

typedef __attribute__((ext_vector_type(16))) _Float16 v16h;
typedef __attribute__((ext_vector_type(8)))  _Float16 v8h;
typedef __attribute__((ext_vector_type(8)))  float    v8f;
typedef __attribute__((ext_vector_type(4)))  int      v4i;

#define BM 128
#define BN 256
#define BK 32
#define LDSP 40   // BK + 8 halfs pad -> 80B row stride, rows stay 16B aligned

#if defined(__HIP_DEVICE_COMPILE__) && __has_builtin(__builtin_amdgcn_global_load_async_to_lds_b128)
#define HAVE_ASYNC 1
#else
#define HAVE_ASYNC 0
#endif

#if HAVE_ASYNC
#define ASYNC_CP_B128(g, l)                                              \
    __builtin_amdgcn_global_load_async_to_lds_b128(                      \
        (__attribute__((address_space(1))) v4i*)(g),                     \
        (__attribute__((address_space(3))) v4i*)(l), 0, 0)
#endif

__device__ __forceinline__ void async_wait0() {
#if HAVE_ASYNC
    asm volatile("s_wait_asynccnt 0x0" ::: "memory");
#endif
}

// ---------------------------------------------------------------- fp32 -> f16
__global__ void k_cvt_f32_f16(const float* __restrict__ in, _Float16* __restrict__ out, int n) {
    int i = blockIdx.x * blockDim.x + threadIdx.x;
    int stride = gridDim.x * blockDim.x;
    for (; i < n; i += stride) out[i] = (_Float16)in[i];
}

// ------------------------------------------- 3x3 weights: OIHW -> tap-major f16
// out[o][tap*Cin + cin] = w[o][cin*9 + tap]
__global__ void k_permw_f32_f16(const float* __restrict__ w, _Float16* __restrict__ out,
                                int Cout, int Cin) {
    int total = Cout * Cin * 9;
    int i = blockIdx.x * blockDim.x + threadIdx.x;
    int stride = gridDim.x * blockDim.x;
    for (; i < total; i += stride) {
        int o   = i / (Cin * 9);
        int r   = i - o * (Cin * 9);
        int cin = r / 9;
        int tap = r - cin * 9;
        out[(size_t)o * (Cin * 9) + tap * Cin + cin] = (_Float16)w[i];
    }
}

// ---------------------------------------------------- softmax rows (in-place, 16384 cols, f16)
__global__ __launch_bounds__(256) void k_softmax16k(_Float16* __restrict__ S) {
    const int cols = 16384;
    __shared__ float red[256];
    const int r = blockIdx.x;
    _Float16* row = S + (size_t)r * cols;
    const int t = threadIdx.x;
    float v[64];
    float mx = -3.0e38f;
    for (int j = 0; j < 64; ++j) { v[j] = (float)row[j * 256 + t]; mx = fmaxf(mx, v[j]); }
    red[t] = mx; __syncthreads();
    for (int s = 128; s > 0; s >>= 1) { if (t < s) red[t] = fmaxf(red[t], red[t + s]); __syncthreads(); }
    mx = red[0]; __syncthreads();
    float sum = 0.f;
    for (int j = 0; j < 64; ++j) { v[j] = __expf(v[j] - mx); sum += v[j]; }
    red[t] = sum; __syncthreads();
    for (int s = 128; s > 0; s >>= 1) { if (t < s) red[t] += red[t + s]; __syncthreads(); }
    const float inv = 1.f / red[0];
    for (int j = 0; j < 64; ++j) row[j * 256 + t] = (_Float16)(v[j] * inv);
}

// -------------------------------------------------------------- WMMA GEMM
// D[M,N] = epilogue( A x B ), A/B f16, f32 accum, double-buffered LDS pipeline.
// Block tile 128x256, 8 waves (2x4), each wave 64x64 = 4x4 WMMA tiles.
//  TA:      A element (m,k) at A[k*lda + m]  (else A[m*lda + k])
//  BMODE 0: B element (k,n) at B[k*ldb + n]
//  BMODE 1: B element (k,n) at B[n*ldb + k]          (async-stageable)
//  BMODE 2: implicit im2col 3x3 conv, H=W=64, tap-major K: k = tap*Cin + cin,
//           Cin = 1<<cinShift, plane stride ldb. One tap per 32-wide K-tile.
// Requires: M % 128 == 0, N % 256 == 0, K % 32 == 0.
template<bool TA, int BMODE>
__global__ __launch_bounds__(256)
void k_gemm_wmma(const _Float16* __restrict__ A, int lda,
                 const _Float16* __restrict__ B, int ldb,
                 _Float16* __restrict__ outh, float* __restrict__ outf, int ldc,
                 int K, int cinShift,
                 const float* __restrict__ gamma, const float* __restrict__ beta,
                 int relu)
{
    __shared__ __align__(16) _Float16 As[2][BM][LDSP];
    __shared__ __align__(16) _Float16 Bs[2][BN][LDSP];

    const int tid  = threadIdx.x;
    const int lane = tid & 31;
    const int wave = tid >> 5;
    const int lr = lane & 15;
    const int lh = lane >> 4;
    const int m0 = blockIdx.y * BM;
    const int n0 = blockIdx.x * BN;
    const int wm = (wave >> 2) * 64;   // 2 waves along M
    const int wn = (wave & 3) * 64;    // 4 waves along N

    // staging index helpers
    const int rowi = tid >> 1;            // 0..127, A contiguous-stage row
    const int segi = (tid & 1) * 16;      // 0 / 16
    const int rowb = tid;                 // 0..255, B transB-stage row (32 halfs)
    const int kli  = tid & 31;            // k lane for scatter/conv staging
    const int chb  = (tid >> 5) * 32;     // 32-wide n chunk for scatter/conv staging

    v8f acc[4][4];
    #pragma unroll
    for (int i = 0; i < 4; ++i)
        #pragma unroll
        for (int j = 0; j < 4; ++j)
            #pragma unroll
            for (int e = 0; e < 8; ++e) acc[i][j][e] = 0.f;

    const int ntiles = K / BK;

    auto fetchA = [&](int kt, int bi, v8h& a0, v8h& a1) {
        if (!TA) {
            const v8h* gp = (const v8h*)(A + (size_t)(m0 + rowi) * lda + kt * BK + segi);
#if HAVE_ASYNC
            _Float16* lp = &As[bi][rowi][segi];
            ASYNC_CP_B128(gp, lp);
            ASYNC_CP_B128(gp + 1, lp + 8);
            (void)a0; (void)a1;
#else
            (void)bi; a0 = gp[0]; a1 = gp[1];
#endif
        } else {
            (void)bi;
            const v8h* gp = (const v8h*)(A + (size_t)(kt * BK + kli) * lda + m0 + (tid >> 5) * 16);
            a0 = gp[0]; a1 = gp[1];
        }
    };
    auto storeA = [&](int bi, v8h a0, v8h a1) {
        if (!TA) {
#if !HAVE_ASYNC
            *(v8h*)&As[bi][rowi][segi]     = a0;
            *(v8h*)&As[bi][rowi][segi + 8] = a1;
#else
            (void)bi; (void)a0; (void)a1;
#endif
        } else {
            const int mc = (tid >> 5) * 16;
            #pragma unroll
            for (int i = 0; i < 8; ++i) As[bi][mc + i][kli] = a0[i];
            #pragma unroll
            for (int i = 0; i < 8; ++i) As[bi][mc + 8 + i][kli] = a1[i];
        }
    };
    auto fetchB = [&](int kt, int bi, v8h* bv, _Float16* fc) {
        if (BMODE == 0) {
            (void)bi; (void)fc;
            const v8h* gp = (const v8h*)(B + (size_t)(kt * BK + kli) * ldb + n0 + chb);
            bv[0] = gp[0]; bv[1] = gp[1]; bv[2] = gp[2]; bv[3] = gp[3];
        } else if (BMODE == 1) {
            const _Float16* gp = B + (size_t)(n0 + rowb) * ldb + kt * BK;
#if HAVE_ASYNC
            (void)bv; (void)fc;
            _Float16* lp = &Bs[bi][rowb][0];
            ASYNC_CP_B128(gp,      lp);
            ASYNC_CP_B128(gp + 8,  lp + 8);
            ASYNC_CP_B128(gp + 16, lp + 16);
            ASYNC_CP_B128(gp + 24, lp + 24);
#else
            (void)bi; (void)fc;
            const v8h* gv = (const v8h*)gp;
            bv[0] = gv[0]; bv[1] = gv[1]; bv[2] = gv[2]; bv[3] = gv[3];
#endif
        } else {
            (void)bi; (void)bv;
            const int kk  = kt * BK + kli;
            const int tap = kk >> cinShift;                  // uniform within K-tile
            const int cin = kk & ((1 << cinShift) - 1);
            const int t3  = tap / 3;                         // tap < 9, cheap
            const int dy  = t3 - 1;
            const int dx  = tap - t3 * 3 - 1;
            const int p   = n0 + chb;
            const int py  = p >> 6, px = p & 63;             // px in {0,32}
            const int yy  = py + dy;
            const bool rowok = (unsigned)yy < 64u;
            const _Float16* plane = B + (size_t)cin * ldb;
            const int base = (yy << 6) + px + dx;
            #pragma unroll
            for (int i = 0; i < 32; ++i) {
                const int xx = px + dx + i;                  // at most one OOB element
                const bool ok = rowok && ((unsigned)xx < 64u);
                int offc = base + i;                         // branchless: clamp + select
                offc = offc < 0 ? 0 : (offc > 4095 ? 4095 : offc);
                const _Float16 ld = plane[offc];
                fc[i] = ok ? ld : (_Float16)0.f;
            }
        }
    };
    auto storeB = [&](int bi, const v8h* bv, const _Float16* fc) {
        if (BMODE == 0) {
            (void)fc;
            #pragma unroll
            for (int c = 0; c < 4; ++c)
                #pragma unroll
                for (int i = 0; i < 8; ++i) Bs[bi][chb + 8 * c + i][kli] = bv[c][i];
        } else if (BMODE == 1) {
#if !HAVE_ASYNC
            (void)fc;
            *(v8h*)&Bs[bi][rowb][0]  = bv[0];
            *(v8h*)&Bs[bi][rowb][8]  = bv[1];
            *(v8h*)&Bs[bi][rowb][16] = bv[2];
            *(v8h*)&Bs[bi][rowb][24] = bv[3];
#else
            (void)bi; (void)bv; (void)fc;
#endif
        } else {
            (void)bv;
            #pragma unroll
            for (int i = 0; i < 32; ++i) Bs[bi][chb + i][kli] = fc[i];
        }
    };
    auto compute = [&](int bi) {
        v16h af[4];
        #pragma unroll
        for (int i = 0; i < 4; ++i) {
            const _Float16* ap = &As[bi][wm + 16 * i + lr][lh * 8];
            v8h alo = *(const v8h*)ap;
            v8h ahi = *(const v8h*)(ap + 16);
            af[i] = __builtin_shufflevector(alo, ahi, 0,1,2,3,4,5,6,7,8,9,10,11,12,13,14,15);
        }
        #pragma unroll
        for (int j = 0; j < 4; ++j) {
            const _Float16* bp = &Bs[bi][wn + 16 * j + lr][lh * 8];
            v8h blo = *(const v8h*)bp;
            v8h bhi = *(const v8h*)(bp + 16);
            v16h bf = __builtin_shufflevector(blo, bhi, 0,1,2,3,4,5,6,7,8,9,10,11,12,13,14,15);
            #pragma unroll
            for (int i = 0; i < 4; ++i)
                acc[i][j] = __builtin_amdgcn_wmma_f32_16x16x32_f16(
                    false, af[i], false, bf, (short)0, acc[i][j], false, false);
        }
    };

    // ---- software pipeline: fetch(t+1) -> compute(t) -> store(t+1), 1 barrier/step
    {
        v8h a0{}, a1{}; v8h bv[4] = {}; _Float16 fcv[32];
        fetchA(0, 0, a0, a1); fetchB(0, 0, bv, fcv);
        storeA(0, a0, a1);    storeB(0, bv, fcv);
        async_wait0();
        __syncthreads();
    }
    for (int t = 0; t < ntiles; ++t) {
        const int cur = t & 1, nxt = cur ^ 1;
        const bool more = (t + 1) < ntiles;
        v8h a0{}, a1{}; v8h bv[4] = {}; _Float16 fcv[32];
        if (more) { fetchA(t + 1, nxt, a0, a1); fetchB(t + 1, nxt, bv, fcv); }
        compute(cur);
        if (more) { storeA(nxt, a0, a1); storeB(nxt, bv, fcv); }
        async_wait0();
        __syncthreads();
    }

    // ---- epilogue: BN affine + ReLU, f16/f32 writes
    #pragma unroll
    for (int i = 0; i < 4; ++i) {
        const int mbase = m0 + wm + 16 * i + 8 * lh;
        #pragma unroll
        for (int r = 0; r < 8; ++r) {
            const int m = mbase + r;
            const float sc = gamma ? gamma[m] : 1.f;
            const float bi = beta ? beta[m] : 0.f;
            #pragma unroll
            for (int j = 0; j < 4; ++j) {
                const int n = n0 + wn + 16 * j + lr;
                float v = acc[i][j][r] * sc + bi;
                if (relu) v = fmaxf(v, 0.f);
                const size_t idx = (size_t)m * ldc + n;
                if (outh) outh[idx] = (_Float16)v;
                if (outf) outf[idx] = v;
            }
        }
    }
}

// ---------------------------------------------------------------- host side
static inline void cvt(const float* in, _Float16* out, int n, hipStream_t s) {
    int blocks = (n + 255) / 256;
    if (blocks > 8192) blocks = 8192;
    k_cvt_f32_f16<<<blocks, 256, 0, s>>>(in, out, n);
}

static inline void permw(const float* in, _Float16* out, int Cout, int Cin, hipStream_t s) {
    int n = Cout * Cin * 9;
    int blocks = (n + 255) / 256;
    if (blocks > 8192) blocks = 8192;
    k_permw_f32_f16<<<blocks, 256, 0, s>>>(in, out, Cout, Cin);
}

static inline void gemm(int mode, bool ta,
                        const _Float16* A, int lda, const _Float16* B, int ldb,
                        _Float16* outh, float* outf, int ldc,
                        int M, int N, int K, int cinShift,
                        const float* g, const float* b, int relu, hipStream_t s)
{
    dim3 grid(N / BN, M / BM), blk(256);
    if (mode == 2)
        k_gemm_wmma<false, 2><<<grid, blk, 0, s>>>(A, lda, B, ldb, outh, outf, ldc, K, cinShift, g, b, relu);
    else if (mode == 1)
        k_gemm_wmma<false, 1><<<grid, blk, 0, s>>>(A, lda, B, ldb, outh, outf, ldc, K, cinShift, g, b, relu);
    else if (ta)
        k_gemm_wmma<true, 0><<<grid, blk, 0, s>>>(A, lda, B, ldb, outh, outf, ldc, K, cinShift, g, b, relu);
    else
        k_gemm_wmma<false, 0><<<grid, blk, 0, s>>>(A, lda, B, ldb, outh, outf, ldc, K, cinShift, g, b, relu);
}

extern "C" void kernel_launch(void* const* d_in, const int* in_sizes, int n_in,
                              void* d_out, int out_size, void* d_ws, size_t ws_size,
                              hipStream_t stream)
{
    (void)in_sizes; (void)n_in; (void)out_size; (void)ws_size;
    const float* x    = (const float*)d_in[0];
    const float* seq  = (const float*)d_in[1];
    const float* wk1  = (const float*)d_in[2];
    const float* gk1  = (const float*)d_in[3];
    const float* bk1  = (const float*)d_in[4];
    const float* wk2  = (const float*)d_in[5];
    const float* gk2  = (const float*)d_in[6];
    const float* bk2  = (const float*)d_in[7];
    const float* wv1  = (const float*)d_in[8];
    const float* gv1  = (const float*)d_in[9];
    const float* bv1  = (const float*)d_in[10];
    const float* wv2  = (const float*)d_in[11];
    const float* gv2  = (const float*)d_in[12];
    const float* bv2  = (const float*)d_in[13];
    const float* wqk1 = (const float*)d_in[14];
    const float* gqk1 = (const float*)d_in[15];
    const float* bqk1 = (const float*)d_in[16];
    const float* wqk2 = (const float*)d_in[17];
    const float* gqk2 = (const float*)d_in[18];
    const float* bqk2 = (const float*)d_in[19];
    const float* wqv1 = (const float*)d_in[20];
    const float* gqv1 = (const float*)d_in[21];
    const float* bqv1 = (const float*)d_in[22];
    const float* wqv2 = (const float*)d_in[23];
    const float* gqv2 = (const float*)d_in[24];
    const float* bqv2 = (const float*)d_in[25];
    const float* wb   = (const float*)d_in[26];
    const float* gb   = (const float*)d_in[27];
    const float* bb   = (const float*)d_in[28];
    float* out = (float*)d_out;

    const int HW = 4096, THW = 16384, Cin = 1024, Ck = 256, Cv = 512, T = 4;

    char* ws = (char*)d_ws;
    size_t off = 0;
    auto halloc = [&](size_t elems) -> _Float16* {
        _Float16* p = (_Float16*)(ws + off);
        off = (off + elems * sizeof(_Float16) + 255) & ~(size_t)255;
        return p;
    };
    _Float16* xh    = halloc((size_t)Cin * HW);
    _Float16* seqh  = halloc((size_t)T * Cin * HW);
    _Float16* wk1h  = halloc((size_t)Ck * Cin);
    _Float16* wk2h  = halloc((size_t)Ck * Ck * 9);
    _Float16* wv1h  = halloc((size_t)Cv * Cin);
    _Float16* wv2h  = halloc((size_t)Cv * Cv * 9);
    _Float16* wqk1h = halloc((size_t)Ck * Cin);
    _Float16* wqk2h = halloc((size_t)Ck * Ck * 9);
    _Float16* wqv1h = halloc((size_t)Cv * Cin);
    _Float16* wqv2h = halloc((size_t)Cv * Cv * 9);
    _Float16* wbh   = halloc((size_t)Cv * 2 * Cv * 9);
    _Float16* tmp   = halloc((size_t)Cv * HW);
    _Float16* mkh   = halloc((size_t)Ck * THW);
    _Float16* mvh   = halloc((size_t)Cv * THW);
    _Float16* qkh   = halloc((size_t)Ck * HW);
    _Float16* cat   = halloc((size_t)2 * Cv * HW);
    _Float16* Sh    = halloc((size_t)HW * THW);

    // conversions (1x1 weights + activations) and tap-major permutes (3x3 weights)
    cvt(x,    xh,    Cin * HW, stream);
    cvt(seq,  seqh,  T * Cin * HW, stream);
    cvt(wk1,  wk1h,  Ck * Cin, stream);
    cvt(wv1,  wv1h,  Cv * Cin, stream);
    cvt(wqk1, wqk1h, Ck * Cin, stream);
    cvt(wqv1, wqv1h, Cv * Cin, stream);
    permw(wk2,  wk2h,  Ck, Ck, stream);
    permw(wv2,  wv2h,  Cv, Cv, stream);
    permw(wqk2, wqk2h, Ck, Ck, stream);
    permw(wqv2, wqv2h, Cv, Cv, stream);
    permw(wb,   wbh,   Cv, 2 * Cv, stream);

    // memory key/value convs over T frames; outputs packed as (C, T*HW) via ldc=THW
    for (int t = 0; t < T; ++t) {
        const _Float16* fr = seqh + (size_t)t * Cin * HW;
        gemm(0, false, wk1h, Cin, fr, HW, tmp, nullptr, HW, Ck, HW, Cin, 0, gk1, bk1, 1, stream);
        gemm(2, false, wk2h, Ck * 9, tmp, HW, mkh + (size_t)t * HW, nullptr, THW, Ck, HW, Ck * 9, 8, gk2, bk2, 1, stream);
        gemm(0, false, wv1h, Cin, fr, HW, tmp, nullptr, HW, Cv, HW, Cin, 0, gv1, bv1, 1, stream);
        gemm(2, false, wv2h, Cv * 9, tmp, HW, mvh + (size_t)t * HW, nullptr, THW, Cv, HW, Cv * 9, 9, gv2, bv2, 1, stream);
    }

    // query key/value convs
    gemm(0, false, wqk1h, Cin, xh, HW, tmp, nullptr, HW, Ck, HW, Cin, 0, gqk1, bqk1, 1, stream);
    gemm(2, false, wqk2h, Ck * 9, tmp, HW, qkh, nullptr, HW, Ck, HW, Ck * 9, 8, gqk2, bqk2, 1, stream);
    gemm(0, false, wqv1h, Cin, xh, HW, tmp, nullptr, HW, Cv, HW, Cin, 0, gqv1, bqv1, 1, stream);
    gemm(2, false, wqv2h, Cv * 9, tmp, HW, cat, nullptr, HW, Cv, HW, Cv * 9, 9, gqv2, bqv2, 1, stream);

    // attention logits: S[q,p] = sum_c qk[c,q]*mk[c,p]
    gemm(0, true, qkh, HW, mkh, THW, Sh, nullptr, THW, HW, THW, Ck, 0, nullptr, nullptr, 0, stream);

    // softmax over THW, in place
    k_softmax16k<<<HW, 256, 0, stream>>>(Sh);

    // mem[c,q] = sum_p mv[c,p]*attn[q,p]  (transB -> fully async-staged GEMM)
    gemm(1, false, mvh, THW, Sh, THW, cat + (size_t)Cv * HW, nullptr, HW, Cv, HW, THW, 0, nullptr, nullptr, 0, stream);

    // final 3x3 conv over concat(qv, mem), affine+ReLU, f32 out
    gemm(2, false, wbh, 2 * Cv * 9, cat, HW, nullptr, out, HW, Cv, HW, 2 * Cv * 9, 10, gb, bb, 1, stream);
}
